// InverseImportanceLinear_64330020159904
// MI455X (gfx1250) — matrix operand order repaired and checked by reading, not compiled, not run on hardware
//
#include <hip/hip_runtime.h>

typedef __attribute__((ext_vector_type(16))) __bf16 v16bf;
typedef __attribute__((ext_vector_type(8)))  __bf16 v8bf;
typedef __attribute__((ext_vector_type(4)))  __bf16 v4bf;
typedef __attribute__((ext_vector_type(8)))  float  v8f;
typedef __attribute__((ext_vector_type(4)))  float  v4f;
typedef __attribute__((ext_vector_type(4)))  int    v4i;

// ---- problem shape (fixed by the reference harness) ----
constexpr int T_DIM = 4096;     // x rows
constexpr int N_DIM = 4096;     // contraction (x cols / Q cols)
constexpr int K_DIM = 11264;    // output cols (Q rows)
constexpr int NG    = 64;       // groups along N (GROUP_SIZE = 64)

// ---- tiling ----
constexpr int BM = 128;         // T rows per block
constexpr int BN = 128;         // K cols per block
constexpr int BK = 64;          // contraction chunk == GROUP_SIZE
constexpr int LDSS = 72;        // LDS row stride (bf16 elems): 144 B, 16B-aligned
constexpr int NTHREADS = 256;   // 8 wave32
constexpr int ABUF = BM * LDSS; // elems per A buffer (9216)
constexpr int BBUF = BN * LDSS;
constexpr int NT = N_DIM / BK;  // 64 k-steps (even)

__global__ __launch_bounds__(NTHREADS) void iil_wmma_gemm(
    const float* __restrict__ x,      // (T, N) f32
    const int*   __restrict__ Q,      // (K, N) int32 codes 0..7
    const float* __restrict__ scales, // (K, NG)
    const float* __restrict__ zeros,  // (K, NG)
    const float* __restrict__ mu1,    // (N)
    const float* __restrict__ mu2,    // (K)
    const float* __restrict__ bias,   // (K)
    float* __restrict__ out)          // (T, K)
{
    __shared__ __bf16 lA[2 * ABUF];
    __shared__ __bf16 lB[2 * BBUF];

    const int tid  = threadIdx.x;
    const int lane = tid & 31;
    const int wave = tid >> 5;
    const int wm   = wave >> 2;   // 0..1 (T dir)
    const int wn   = wave & 3;    // 0..3 (K dir)
    const int m0   = blockIdx.y * BM;
    const int k0   = blockIdx.x * BN;

    // loader geometry: fixed 4-col slice, 8 rows with stride 16
    const int lcol4 = (tid & 15) * 4;
    const int lrow0 = tid >> 4;

    // global cursors; row offsets (i*16 rows) become instruction immediates
    const float* xPtr  = x + (size_t)(m0 + lrow0) * N_DIM + lcol4;
    const int*   qPtr  = Q + (size_t)(k0 + lrow0) * N_DIM + lcol4;
    const float* sPtr  = scales + (size_t)(k0 + lrow0) * NG;
    const float* zPtr  = zeros  + (size_t)(k0 + lrow0) * NG;
    const float* muPtr = mu1 + lcol4;

    float m2[8];
#pragma unroll
    for (int i = 0; i < 8; ++i) m2[i] = mu2[k0 + lrow0 + 16 * i];

    v8f acc[4][2];
#pragma unroll
    for (int mi = 0; mi < 4; ++mi)
#pragma unroll
        for (int ni = 0; ni < 2; ++ni)
            acc[mi][ni] = (v8f){0.f, 0.f, 0.f, 0.f, 0.f, 0.f, 0.f, 0.f};

    // staging registers (global -> regs -> LDS)
    v4f xa[8];
    v4i qb[8];
    v4f muv;
    float sv[8], zv[8];

    auto loadStage = [&]() {
#pragma unroll
        for (int i = 0; i < 8; ++i) {
            xa[i] = *(const v4f*)(xPtr + (size_t)i * 16 * N_DIM);
            qb[i] = *(const v4i*)(qPtr + (size_t)i * 16 * N_DIM);
            sv[i] = sPtr[(size_t)i * 16 * NG];
            zv[i] = zPtr[(size_t)i * 16 * NG];
        }
        muv = *(const v4f*)muPtr;
        xPtr += BK; qPtr += BK; muPtr += BK;
        sPtr += 1;  zPtr += 1;
    };

    __bf16* aDst = &lA[lrow0 * LDSS + lcol4];
    __bf16* bDst = &lB[lrow0 * LDSS + lcol4];

    auto storeStage = [&](int buf) {  // called with literal buf -> DS immediates
#pragma unroll
        for (int i = 0; i < 8; ++i) {
            // A: f32 -> bf16
            v4bf av;
            av.x = (__bf16)xa[i].x;  av.y = (__bf16)xa[i].y;
            av.z = (__bf16)xa[i].z;  av.w = (__bf16)xa[i].w;
            *(v4bf*)(aDst + buf * ABUF + i * 16 * LDSS) = av;
            // B: fma(qf, s*mu2, -z*s*mu2) * mu1 -> bf16
            const float sm   = sv[i] * m2[i];
            const float nzsm = -zv[i] * sm;
            v4bf bv;
            bv.x = (__bf16)(__builtin_fmaf((float)qb[i].x, sm, nzsm) * muv.x);
            bv.y = (__bf16)(__builtin_fmaf((float)qb[i].y, sm, nzsm) * muv.y);
            bv.z = (__bf16)(__builtin_fmaf((float)qb[i].z, sm, nzsm) * muv.z);
            bv.w = (__bf16)(__builtin_fmaf((float)qb[i].w, sm, nzsm) * muv.w);
            *(v4bf*)(bDst + buf * BBUF + i * 16 * LDSS) = bv;
        }
    };

    // fragment base addresses (ISA VGPR layouts for 16-bit A 16x32 / B 32x16)
    const int arow  = wm * 64 + (lane & 15);
    const int brow  = wn * 32 + (lane & 15);
    const int ahalf = (lane < 16) ? 0 : 8;
    const int bhalf = (lane < 16) ? 0 : 8;
    const __bf16* aFragBase = &lA[arow * LDSS + ahalf];
    const __bf16* bFragBase = &lB[brow * LDSS + bhalf];

    auto computeStage = [&](int buf) {
#pragma unroll
        for (int kk = 0; kk < BK; kk += 32) {
            v16bf af[4], bfr[2];
#pragma unroll
            for (int mi = 0; mi < 4; ++mi) {
                const __bf16* p = aFragBase + buf * ABUF + mi * 16 * LDSS + kk;
                v8bf lo = *(const v8bf*)p;         // K = half .. half+7
                v8bf hi = *(const v8bf*)(p + 16);  // K = half+16 .. half+23
                af[mi] = __builtin_shufflevector(lo, hi,
                         0,1,2,3,4,5,6,7,8,9,10,11,12,13,14,15);
            }
#pragma unroll
            for (int ni = 0; ni < 2; ++ni) {
                const __bf16* p = bFragBase + buf * BBUF + ni * 16 * LDSS + kk;
                v8bf lo = *(const v8bf*)p;         // K = half .. half+7
                v8bf hi = *(const v8bf*)(p + 16);  // K = half+16 .. half+23
                bfr[ni] = __builtin_shufflevector(lo, hi,
                         0,1,2,3,4,5,6,7,8,9,10,11,12,13,14,15);
            }
#pragma unroll
            for (int mi = 0; mi < 4; ++mi)
#pragma unroll
                for (int ni = 0; ni < 2; ++ni)
                    acc[mi][ni] = __builtin_amdgcn_wmma_f32_16x16x32_bf16(
                        false, af[mi], false, bfr[ni],
                        (short)0, acc[mi][ni], false, false);
        }
    };

    // ---- pipelined main loop, manually 2x unrolled so buf is compile-time ----
    loadStage();
    storeStage(0);
    __syncthreads();
#pragma unroll 1
    for (int kt = 0; kt < NT; kt += 2) {
        if (kt + 1 < NT) loadStage();     // next tile's global loads in flight
        computeStage(0);                  // WMMAs overlap the loads
        if (kt + 1 < NT) storeStage(1);
        __syncthreads();
        if (kt + 2 < NT) loadStage();
        computeStage(1);
        if (kt + 2 < NT) storeStage(0);
        __syncthreads();
    }

    // ---- epilogue: D layout -> out(T,K), add bias; NT stores keep L2 for x/Q ----
    const int colBase = k0 + wn * 32 + (lane & 15);
    const int rowBase = m0 + wm * 64 + ((lane < 16) ? 0 : 8);
    float* oPtr = out + (size_t)rowBase * K_DIM + colBase;
#pragma unroll
    for (int ni = 0; ni < 2; ++ni) {
        const float b = bias[colBase + 16 * ni];
#pragma unroll
        for (int mi = 0; mi < 4; ++mi) {
#pragma unroll
            for (int v = 0; v < 8; ++v) {
                __builtin_nontemporal_store(
                    acc[mi][ni][v] + b,
                    oPtr + (size_t)(16 * mi + v) * K_DIM + 16 * ni);
            }
        }
    }
}

extern "C" void kernel_launch(void* const* d_in, const int* in_sizes, int n_in,
                              void* d_out, int out_size, void* d_ws, size_t ws_size,
                              hipStream_t stream) {
    const float* x      = (const float*)d_in[0];
    const int*   Q      = (const int*)  d_in[1];
    const float* scales = (const float*)d_in[2];
    const float* zeros  = (const float*)d_in[3];
    const float* mu1    = (const float*)d_in[4];
    const float* mu2    = (const float*)d_in[5];
    const float* bias   = (const float*)d_in[6];
    float* out = (float*)d_out;

    // Reference shapes: T=4096 (32x128), K=11264 (88x128), N=4096 (64x64)
    dim3 grid(K_DIM / BN, T_DIM / BM);
    iil_wmma_gemm<<<grid, NTHREADS, 0, stream>>>(
        x, Q, scales, zeros, mu1, mu2, bias, out);
}
